// RNN_75385265979616
// MI455X (gfx1250) — compile-verified, matrix operand order checked by reading, not run on hardware
//
#include <hip/hip_runtime.h>
#include <cmath>

#define B_   128
#define T_   512
#define IN_  64
#define H_   1024
#define OUT_ 64
#define ALPHA 0.1f

#define NGROUP 8          // batch groups, 16 rows each
#define WPG    9          // 8 H-slice WGs + 1 pred WG per group
#define MB     16         // batch rows per group (one WMMA M tile)
#define COLS   128        // H columns owned per H-WG
#define KTOT   1088       // 1024 (fr) + 64 (x_t) unified K
#define KPAD   1096       // padded LDS row stride in halves (bank-conflict avoidance)

typedef __attribute__((ext_vector_type(16))) _Float16 v16h;
typedef __attribute__((ext_vector_type(8)))  _Float16 v8h;
typedef __attribute__((ext_vector_type(4)))  float    v4f;
typedef __attribute__((ext_vector_type(8)))  float    v8f;

__global__ void rnn_init(const float* __restrict__ fr0, float* frbuf,
                         _Float16* fr16buf, unsigned* cnt) {
  unsigned i = blockIdx.x * blockDim.x + threadIdx.x;
  if (i < (unsigned)(B_ * H_)) {
    float v = fr0[i];
    frbuf[i]   = v;                 // fp32 ping buffer 0 = fr_0 (exact state)
    fr16buf[i] = (_Float16)v;       // f16 mirror (WMMA A operand source)
  }
  if (i < NGROUP) cnt[i * 32] = 0u; // per-group barrier counters
}

__global__ void __launch_bounds__(256, 1)
rnn_persist(const float* __restrict__ inp,     const float* __restrict__ noise,
            const float* __restrict__ W_in_w,  const float* __restrict__ W_in_b,
            const float* __restrict__ W_rec_w, const float* __restrict__ W_rec_b,
            const float* __restrict__ W_out_w, const float* __restrict__ W_out_b,
            float* __restrict__ frbuf, _Float16* __restrict__ fr16buf,
            unsigned* __restrict__ cnt,
            float* __restrict__ preds, float* __restrict__ rates)
{
  // 280,576 B weights + 35,072 B activations = 315,648 B  (< 320 KB WGP LDS)
  __shared__ _Float16 Wl[COLS * KPAD];
  __shared__ _Float16 Al[MB * KPAD];

  const int tid  = threadIdx.x;
  const int wave = tid >> 5;
  const int lane = tid & 31;
  const int g    = blockIdx.x / WPG;        // batch group
  const int role = blockIdx.x % WPG;        // 0..7 = H slice, 8 = pred
  const bool isPred = (role == 8);
  const int b0 = g * MB;

  // ---- one-time: stage this WG's weight slice as f16 into LDS (read from L2 once) ----
  const int ncols = isPred ? OUT_ : COLS;
  for (int i = tid; i < ncols * KTOT; i += 256) {
    int c = i / KTOT, k = i - c * KTOT;
    float w;
    if (!isPred) {
      int h = role * COLS + c;
      w = (k < H_) ? W_rec_w[(size_t)h * H_ + k] : W_in_w[h * IN_ + (k - H_)];
    } else {
      w = (k < H_) ? W_out_w[(size_t)c * H_ + k] : 0.0f;   // x_t contributes 0 to pred
    }
    Wl[c * KPAD + k] = (_Float16)w;
  }

  // per-lane constants (wave owns one 16x16 output tile)
  const int ln   = lane & 15;
  const int loff = (lane >> 4) << 3;                        // K sub-offset: 0 or 8
  const bool active = !(isPred && wave >= 4);               // pred WG needs only 4 tiles
  int col = 0; float bias_c = 0.0f;
  if (active) {
    if (!isPred) { col = role * COLS + wave * 16 + ln; bias_c = W_in_b[col] + W_rec_b[col]; }
    else         { col = wave * 16 + ln;               bias_c = W_out_b[col]; }
  }
  const int ksteps = isPred ? (H_ / 32) : (KTOT / 32);
  const _Float16* Arow = Al + ln * KPAD + loff;
  const _Float16* Brow = Wl + (wave * 16 + ln) * KPAD + loff;
  unsigned* gcnt = cnt + g * 32;

  // async-staging geometry: 16 rows x 2048B of fr -> 2048 x 16B chunks, 8 per thread
  const unsigned AlBase = (unsigned)(uintptr_t)(&Al[0]);    // AS3 offset of A tile

  for (int t = 0; t < T_; ++t) {
    const float*    frcur   = frbuf   + (size_t)(t & 1) * (B_ * H_);
    float*          frnxt   = frbuf   + (size_t)((t + 1) & 1) * (B_ * H_);
    const _Float16* fr16cur = fr16buf + (size_t)(t & 1) * (B_ * H_);
    _Float16*       fr16nxt = fr16buf + (size_t)((t + 1) & 1) * (B_ * H_);

    // ---- stage A[fr part] via async memory->LDS DMA (no VGPR/VALU transit) ----
    #pragma unroll
    for (int r = 0; r < 8; ++r) {
      int i = tid + 256 * r;            // chunk id 0..2047
      int m = i >> 7;                   // batch row (128 chunks of 16B per row)
      int c = i & 127;                  // 16B chunk within row
      unsigned lds_off = AlBase + (unsigned)(m * KPAD + c * 8) * 2u;
      const _Float16* gp = fr16cur + (size_t)(b0 + m) * H_ + c * 8;
      asm volatile("global_load_async_to_lds_b128 %0, %1, off"
                   :: "v"(lds_off), "v"(gp) : "memory");
    }
    // ---- stage A[x_t part] manually (16x64 fp32 -> f16), overlaps with async DMA ----
    {
      int i  = tid * 4;                 // 4 halves per thread, 1024 total
      int m  = i >> 6;
      int kx = i & 63;
      v4f xv = *(const v4f*)(inp + ((size_t)(b0 + m) * T_ + t) * IN_ + kx);
      _Float16* dp = Al + m * KPAD + H_ + kx;
      dp[0] = (_Float16)xv.x; dp[1] = (_Float16)xv.y;
      dp[2] = (_Float16)xv.z; dp[3] = (_Float16)xv.w;
    }
    asm volatile("s_wait_asynccnt 0x0" ::: "memory");   // own async copies done
    __syncthreads();                                    // all waves' staging visible

    if (active) {
      v8f acc = {};
      for (int kk = 0; kk < ksteps; ++kk) {
        const _Float16* ap = Arow + kk * 32;
        const _Float16* bp = Brow + kk * 32;
        v8h a0 = *(const v8h*)(ap);
        v8h a1 = *(const v8h*)(ap + 16);
        v8h c0 = *(const v8h*)(bp);
        v8h c1 = *(const v8h*)(bp + 16);
        v16h a = __builtin_shufflevector(a0, a1, 0,1,2,3,4,5,6,7,8,9,10,11,12,13,14,15);
        v16h b = __builtin_shufflevector(c0, c1, 0,1,2,3,4,5,6,7,8,9,10,11,12,13,14,15);
        acc = __builtin_amdgcn_wmma_f32_16x16x32_f16(false, a, false, b,
                                                     (short)0, acc, false, false);
      }
      if (!isPred) {
        #pragma unroll
        for (int r = 0; r < 8; ++r) {
          int m = r + loff;
          int b = b0 + m;
          float z  = tanhf(acc[r] + bias_c + noise[((size_t)t * B_ + b) * H_ + col]);
          float fo = frcur[(size_t)b * H_ + col];
          float fn = (1.0f - ALPHA) * fo + ALPHA * z;
          frnxt[(size_t)b * H_ + col] = fn;                 // exact fp32 state
          fr16nxt[(size_t)b * H_ + col] = (_Float16)fn;     // f16 mirror for next A
          rates[((size_t)b * T_ + t) * H_ + col] = fn;
        }
      } else {
        #pragma unroll
        for (int r = 0; r < 8; ++r) {
          int m = r + loff;
          int b = b0 + m;
          float s = acc[r] + bias_c;                        // pred uses fr BEFORE update
          preds[((size_t)b * T_ + t) * OUT_ + col] = 1.0f / (1.0f + __expf(-s));
        }
      }
    }

    // ---- per-group barrier (ping-pong => one barrier per step suffices) ----
    __threadfence();                 // make fr_nxt / fr16_nxt writes agent-visible
    __syncthreads();                 // all waves of this WG done with step t
    if (tid == 0) {
      __hip_atomic_fetch_add(gcnt, 1u, __ATOMIC_RELEASE, __HIP_MEMORY_SCOPE_AGENT);
      unsigned target = (unsigned)WPG * (unsigned)(t + 1);
      while (__hip_atomic_load(gcnt, __ATOMIC_ACQUIRE, __HIP_MEMORY_SCOPE_AGENT) < target)
        __builtin_amdgcn_s_sleep(1);
    }
    __syncthreads();
    __threadfence();                 // invalidate stale cached fr before next staging
  }
}

extern "C" void kernel_launch(void* const* d_in, const int* in_sizes, int n_in,
                              void* d_out, int out_size, void* d_ws, size_t ws_size,
                              hipStream_t stream) {
  (void)in_sizes; (void)n_in; (void)out_size; (void)ws_size;
  const float* inp     = (const float*)d_in[0];
  const float* fr0     = (const float*)d_in[1];
  const float* noise   = (const float*)d_in[2];
  const float* W_in_w  = (const float*)d_in[3];
  const float* W_in_b  = (const float*)d_in[4];
  const float* W_rec_w = (const float*)d_in[5];
  const float* W_rec_b = (const float*)d_in[6];
  const float* W_out_w = (const float*)d_in[7];
  const float* W_out_b = (const float*)d_in[8];

  float* preds = (float*)d_out;                              // [B,T,OUT] first
  float* rates = preds + (size_t)B_ * T_ * OUT_;             // then [B,T,H]

  char* ws = (char*)d_ws;
  float*    frbuf   = (float*)ws;                            // 2*128*1024 fp32 = 1 MB
  _Float16* fr16buf = (_Float16*)(ws + sizeof(float) * 2u * B_ * H_);   // 0.5 MB
  unsigned* cnt     = (unsigned*)(ws + sizeof(float) * 2u * B_ * H_
                                     + sizeof(_Float16) * 2u * B_ * H_);

  rnn_init<<<(B_ * H_ + 255) / 256, 256, 0, stream>>>(fr0, frbuf, fr16buf, cnt);
  rnn_persist<<<NGROUP * WPG, 256, 0, stream>>>(inp, noise, W_in_w, W_in_b,
                                                W_rec_w, W_rec_b, W_out_w, W_out_b,
                                                frbuf, fr16buf, cnt, preds, rates);
}
